// SaladHead_84920093377044
// MI455X (gfx1250) — compile-verified, matrix operand order checked by reading, not run on hardware
//
#include <hip/hip_runtime.h>
#include <hip/hip_bf16.h>
#include <math.h>

// ---------------------------------------------------------------------------
// Problem constants (from reference setup_inputs): B=4, S=4, N=1029, C=2048,
// patch_start_idx=5 -> n = 4*1024 = 4096 patch tokens per batch, m = 64.
// ---------------------------------------------------------------------------
#define BB   4
#define SS   4
#define NTOK 1029
#define CC   2048
#define PSI  5
#define NPAT 4096            // S * (NTOK - PSI)
#define MROWS (BB * NPAT)    // 16384
#define MCLUS 64
#define OUTROW 8704          // 128*64 + 256 + 256

typedef __attribute__((ext_vector_type(16))) __bf16 v16bf;
typedef __attribute__((ext_vector_type(8)))  float  v8f;

__device__ __forceinline__ unsigned short f2bf(float x) {
    unsigned u = __float_as_uint(x);
    unsigned r = u + 0x7FFFu + ((u >> 16) & 1u);   // round-to-nearest-even
    return (unsigned short)(r >> 16);
}

union FragBF { v16bf v; uint4 q[2]; };

// Load a 16x32 bf16 fragment (A layout; B uses same pattern on Wt[N,K]).
// Lane L: row = r0 + (L&15); halves: K = k0+(L>>4)*8 .. +8 and K+16 .. +8.
__device__ __forceinline__ v16bf load_frag(const unsigned short* base, int ld,
                                           int r0, int k0, int lane) {
    int r  = r0 + (lane & 15);
    int kk = k0 + ((lane >> 4) << 3);
    const unsigned short* p = base + (size_t)r * ld + kk;
    FragBF f;
    f.q[0] = *reinterpret_cast<const uint4*>(p);
    f.q[1] = *reinterpret_cast<const uint4*>(p + 16);
    return f.v;
}

// ---------------------------------------------------------------------------
// WMMA GEMM: Out[M,N] = act(A[M,K] * Wt[N,K]^T + bias). 8 waves/block, each
// wave computes a 32x64 tile = 8 v_wmma_f32_16x16x32_bf16 per K-step.
// ---------------------------------------------------------------------------
template<bool RELU, bool OUT_BF16>
__global__ void gemm_bf16_wmma(const unsigned short* __restrict__ A, int lda,
                               const unsigned short* __restrict__ Wt, int ldw,
                               void* __restrict__ Out, int ldo,
                               int M, int N, int K,
                               const float* __restrict__ bias0,
                               const float* __restrict__ bias1, int bsplit) {
    const int lane = threadIdx.x & 31;
    const int wave = threadIdx.x >> 5;
    const int ntn  = N >> 6;
    const int tile = blockIdx.x * 8 + wave;
    const int total = (M >> 5) * ntn;
    if (tile >= total) return;
    const int m0 = (tile / ntn) << 5;
    const int n0 = (tile % ntn) << 6;

    v8f acc[2][4];
#pragma unroll
    for (int i = 0; i < 2; ++i)
#pragma unroll
        for (int j = 0; j < 4; ++j)
#pragma unroll
            for (int r = 0; r < 8; ++r) acc[i][j][r] = 0.0f;

    for (int k0 = 0; k0 < K; k0 += 32) {
        v16bf a0 = load_frag(A, lda, m0,      k0, lane);
        v16bf a1 = load_frag(A, lda, m0 + 16, k0, lane);
        v16bf b0 = load_frag(Wt, ldw, n0,      k0, lane);
        v16bf b1 = load_frag(Wt, ldw, n0 + 16, k0, lane);
        v16bf b2 = load_frag(Wt, ldw, n0 + 32, k0, lane);
        v16bf b3 = load_frag(Wt, ldw, n0 + 48, k0, lane);
        acc[0][0] = __builtin_amdgcn_wmma_f32_16x16x32_bf16(false, a0, false, b0, (short)0, acc[0][0], false, false);
        acc[0][1] = __builtin_amdgcn_wmma_f32_16x16x32_bf16(false, a0, false, b1, (short)0, acc[0][1], false, false);
        acc[0][2] = __builtin_amdgcn_wmma_f32_16x16x32_bf16(false, a0, false, b2, (short)0, acc[0][2], false, false);
        acc[0][3] = __builtin_amdgcn_wmma_f32_16x16x32_bf16(false, a0, false, b3, (short)0, acc[0][3], false, false);
        acc[1][0] = __builtin_amdgcn_wmma_f32_16x16x32_bf16(false, a1, false, b0, (short)0, acc[1][0], false, false);
        acc[1][1] = __builtin_amdgcn_wmma_f32_16x16x32_bf16(false, a1, false, b1, (short)0, acc[1][1], false, false);
        acc[1][2] = __builtin_amdgcn_wmma_f32_16x16x32_bf16(false, a1, false, b2, (short)0, acc[1][2], false, false);
        acc[1][3] = __builtin_amdgcn_wmma_f32_16x16x32_bf16(false, a1, false, b3, (short)0, acc[1][3], false, false);
    }

    // C/D layout: lane L -> col N = L&15, rows M = 8*(L>>4) + r
#pragma unroll
    for (int mi = 0; mi < 2; ++mi)
#pragma unroll
        for (int ni = 0; ni < 4; ++ni) {
            int col = n0 + (ni << 4) + (lane & 15);
            float bias = (col < bsplit) ? bias0[col] : bias1[col - bsplit];
            int rowb = m0 + (mi << 4) + ((lane >> 4) << 3);
#pragma unroll
            for (int r = 0; r < 8; ++r) {
                float vv = acc[mi][ni][r] + bias;
                if (RELU) vv = fmaxf(vv, 0.0f);
                size_t off = (size_t)(rowb + r) * ldo + col;
                if (OUT_BF16) ((unsigned short*)Out)[off] = f2bf(vv);
                else          ((float*)Out)[off] = vv;
            }
        }
}

// ---------------------------------------------------------------------------
// Pack patch tokens (tokens[:,:,5:,:]) -> bf16 [16384, 2048]
// ---------------------------------------------------------------------------
__global__ void pack_patch(const float* __restrict__ tokens,
                           unsigned short* __restrict__ A) {
    size_t idx = ((size_t)blockIdx.x * blockDim.x + threadIdx.x) * 8;
    if (idx >= (size_t)MROWS * CC) return;
    size_t row = idx >> 11;
    int c = (int)(idx & 2047);
    int t = (int)(row & 1023);
    int s = (int)((row >> 10) & 3);
    int b = (int)(row >> 12);
    const float* src = tokens + (((size_t)(b * SS + s) * NTOK) + PSI + t) * CC + c;
    float4 x0 = *(const float4*)src;
    float4 x1 = *(const float4*)(src + 4);
    union { unsigned short u[8]; uint4 q; } o;
    o.u[0] = f2bf(x0.x); o.u[1] = f2bf(x0.y); o.u[2] = f2bf(x0.z); o.u[3] = f2bf(x0.w);
    o.u[4] = f2bf(x1.x); o.u[5] = f2bf(x1.y); o.u[6] = f2bf(x1.z); o.u[7] = f2bf(x1.w);
    *(uint4*)(A + idx) = o.q;
}

// W[K,N] fp32 -> Wt[N,K] bf16
__global__ void transpose_w_bf16(const float* __restrict__ W,
                                 unsigned short* __restrict__ Wt, int K, int N) {
    int idx = blockIdx.x * blockDim.x + threadIdx.x;
    if (idx >= K * N) return;
    int n = idx / K, k = idx - n * K;
    Wt[idx] = f2bf(W[(size_t)k * N + n]);
}

// Gather camera / register tokens: rows = B*S*ntok, each row 2048 floats
__global__ void gather_tokens(const float* __restrict__ tokens,
                              float* __restrict__ out, int first, int ntok) {
    int idx = blockIdx.x * blockDim.x + threadIdx.x;
    int rows = BB * SS * ntok;
    if (idx >= rows * CC) return;
    int r = idx >> 11, c = idx & 2047;
    int t = r % ntok;
    int s = (r / ntok) % SS;
    int b = r / (ntok * SS);
    out[idx] = tokens[(((size_t)(b * SS + s) * NTOK) + first + t) * CC + c];
}

// Small fp32 linear: Y[M,N] = act(X[M,K] @ W[K,N] + b)
__global__ void linear_f32(const float* __restrict__ X, const float* __restrict__ W,
                           const float* __restrict__ bias, float* __restrict__ Y,
                           int M, int N, int K, int relu) {
    int idx = blockIdx.x * blockDim.x + threadIdx.x;
    if (idx >= M * N) return;
    int m = idx / N, n = idx - m * N;
    float acc = bias[n];
    const float* xr = X + (size_t)m * K;
    for (int k = 0; k < K; ++k) acc += xr[k] * W[(size_t)k * N + n];
    if (relu) acc = fmaxf(acc, 0.0f);
    Y[idx] = acc;
}

// Row L2 norm: Y[row*ldy + yoff + c] = X[row*N + c] / max(||row||,1e-12)
__global__ void l2norm_rows(const float* __restrict__ X, float* __restrict__ Y,
                            int N, int ldy, int yoff) {
    int row = blockIdx.x;
    __shared__ float red[256];
    float s = 0.0f;
    for (int c = threadIdx.x; c < N; c += blockDim.x) {
        float v = X[(size_t)row * N + c];
        s += v * v;
    }
    red[threadIdx.x] = s;
    __syncthreads();
    for (int off = 128; off; off >>= 1) {
        if ((int)threadIdx.x < off) red[threadIdx.x] += red[threadIdx.x + off];
        __syncthreads();
    }
    float nrm = fmaxf(sqrtf(red[0]), 1e-12f);
    for (int c = threadIdx.x; c < N; c += blockDim.x)
        Y[(size_t)row * ldy + yoff + c] = X[(size_t)row * N + c] / nrm;
}

// GeM pooling over L rows per batch: G[b,c] = (mean_l max(H,eps)^p)^(1/p)
__global__ void gem_pool(const float* __restrict__ H, float* __restrict__ G,
                         int L, int C, const float* __restrict__ pw) {
    int idx = blockIdx.x * blockDim.x + threadIdx.x;
    if (idx >= BB * C) return;
    int b = idx / C, c = idx - b * C;
    float p = pw[0];
    float acc = 0.0f;
    for (int l = 0; l < L; ++l) {
        float v = fmaxf(H[(size_t)(b * L + l) * C + c], 1e-6f);
        acc += powf(v, p);
    }
    G[idx] = powf(acc / (float)L, 1.0f / p);
}

// ---------------------------------------------------------------------------
// Log-domain Sinkhorn, one block per batch. scores: [B*4096, 64] fp32.
// Writes P[b,k,j] = exp(M + u + v - norm) for k < 64.
// ---------------------------------------------------------------------------
__global__ void __launch_bounds__(1024)
sinkhorn_kernel(const float* __restrict__ scores, const float* __restrict__ dustp,
                float* __restrict__ P) {
    const int bb = blockIdx.x;
    const int tid = threadIdx.x;
    const int lane = tid & 31, wid = tid >> 5;
    __shared__ float u[MCLUS + 1];
    __shared__ float v[NPAT];
    const float dust = dustp[0];
    const float norm = -logf((float)(NPAT + MCLUS));          // -log(4160)
    const float la_last = norm + logf((float)(NPAT - MCLUS)); // +log(4032)
    const float* sc = scores + (size_t)bb * NPAT * MCLUS;

    for (int j = tid; j < NPAT; j += 1024) v[j] = 0.0f;
    if (tid <= MCLUS) u[tid] = 0.0f;
    __syncthreads();

    for (int it = 0; it < 3; ++it) {
        // u[k] = log_a[k] - lse_j(M[k,j] + v[j]) ; one wave per k
        for (int k = wid; k <= MCLUS; k += 32) {
            float mx = -1e30f, s = 0.0f;
            for (int j = lane; j < NPAT; j += 32) {
                float Mkj = (k < MCLUS) ? sc[(size_t)j * MCLUS + k] : dust;
                float x = Mkj + v[j];
                if (x > mx) { s = s * __expf(mx - x) + 1.0f; mx = x; }
                else        { s += __expf(x - mx); }
            }
            for (int off = 16; off; off >>= 1) {
                float mo = __shfl_xor(mx, off, 32);
                float so = __shfl_xor(s,  off, 32);
                float m2 = fmaxf(mx, mo);
                s = s * __expf(mx - m2) + so * __expf(mo - m2);
                mx = m2;
            }
            if (lane == 0) {
                float la = (k == MCLUS) ? la_last : norm;
                u[k] = la - (mx + __logf(s));
            }
        }
        __syncthreads();
        // v[j] = log_b - lse_k(M[k,j] + u[k])
        for (int j = tid; j < NPAT; j += 1024) {
            float mx = -1e30f, s = 0.0f;
            for (int k = 0; k <= MCLUS; ++k) {
                float Mkj = (k < MCLUS) ? sc[(size_t)j * MCLUS + k] : dust;
                float x = Mkj + u[k];
                if (x > mx) { s = s * __expf(mx - x) + 1.0f; mx = x; }
                else        { s += __expf(x - mx); }
            }
            v[j] = norm - (mx + __logf(s));
        }
        __syncthreads();
    }

    float* Pb = P + (size_t)bb * MCLUS * NPAT;
    for (int idx = tid; idx < MCLUS * NPAT; idx += 1024) {
        int k = idx >> 12, j = idx & (NPAT - 1);
        Pb[idx] = __expf(sc[(size_t)j * MCLUS + k] + u[k] + v[j] - norm);
    }
}

// ---------------------------------------------------------------------------
// pooled[b,c,k] = sum_j feats[b,j,c]*P[b,k,j]; L2-normalize over c; write out.
// grid = B*64 blocks (b,k); 128 threads (c).
// ---------------------------------------------------------------------------
__global__ void pooled_norm(const float* __restrict__ feats,
                            const float* __restrict__ P, float* __restrict__ out) {
    int b = blockIdx.x >> 6, k = blockIdx.x & 63, c = threadIdx.x;
    __shared__ float pj[128];
    __shared__ float red[128];
    const float* fb = feats + (size_t)b * NPAT * 128;
    const float* pb = P + ((size_t)b * MCLUS + k) * NPAT;
    float acc = 0.0f;
    for (int j0 = 0; j0 < NPAT; j0 += 128) {
        __syncthreads();
        pj[c] = pb[j0 + c];
        __syncthreads();
#pragma unroll 8
        for (int jj = 0; jj < 128; ++jj)
            acc += fb[(size_t)(j0 + jj) * 128 + c] * pj[jj];
    }
    red[c] = acc * acc;
    __syncthreads();
    for (int off = 64; off; off >>= 1) {
        if (c < off) red[c] += red[c + off];
        __syncthreads();
    }
    float nrm = fmaxf(sqrtf(red[0]), 1e-12f);
    out[(size_t)b * OUTROW + c * 64 + k] = acc / nrm;
}

// ---------------------------------------------------------------------------
extern "C" void kernel_launch(void* const* d_in, const int* in_sizes, int n_in,
                              void* d_out, int out_size, void* d_ws, size_t ws_size,
                              hipStream_t stream) {
    (void)in_sizes; (void)n_in; (void)out_size; (void)ws_size;
    const float* tokens = (const float*)d_in[0];
    const float* cam_w1 = (const float*)d_in[2];  const float* cam_b1 = (const float*)d_in[3];
    const float* cam_w2 = (const float*)d_in[4];  const float* cam_b2 = (const float*)d_in[5];
    const float* cam_p  = (const float*)d_in[6];
    const float* cam_w3 = (const float*)d_in[7];  const float* cam_b3 = (const float*)d_in[8];
    const float* cam_w4 = (const float*)d_in[9];  const float* cam_b4 = (const float*)d_in[10];
    const float* reg_w1 = (const float*)d_in[11]; const float* reg_b1 = (const float*)d_in[12];
    const float* reg_w2 = (const float*)d_in[13]; const float* reg_b2 = (const float*)d_in[14];
    const float* reg_p  = (const float*)d_in[15];
    const float* reg_w3 = (const float*)d_in[16]; const float* reg_b3 = (const float*)d_in[17];
    const float* reg_w4 = (const float*)d_in[18]; const float* reg_b4 = (const float*)d_in[19];
    const float* pf_w1  = (const float*)d_in[20]; const float* pf_b1  = (const float*)d_in[21];
    const float* pf_w2  = (const float*)d_in[22]; const float* pf_b2  = (const float*)d_in[23];
    const float* ps_w1  = (const float*)d_in[24]; const float* ps_b1  = (const float*)d_in[25];
    const float* ps_w2  = (const float*)d_in[26]; const float* ps_b2  = (const float*)d_in[27];
    const float* dust   = (const float*)d_in[28];
    float* out = (float*)d_out;

    // workspace carve-up
    char* ws = (char*)d_ws;
    size_t off = 0;
    auto alloc = [&](size_t bytes) { size_t r = off; off = (off + bytes + 255) & ~(size_t)255; return r; };
    unsigned short* A_bf  = (unsigned short*)(ws + alloc((size_t)MROWS * CC * 2));
    unsigned short* W1t   = (unsigned short*)(ws + alloc((size_t)1024 * 2048 * 2));
    unsigned short* Hbf   = (unsigned short*)(ws + alloc((size_t)MROWS * 1024 * 2));
    unsigned short* W2ft  = (unsigned short*)(ws + alloc((size_t)128 * 512 * 2));
    unsigned short* W2st  = (unsigned short*)(ws + alloc((size_t)64 * 512 * 2));
    float* feats  = (float*)(ws + alloc((size_t)MROWS * 128 * 4));
    float* scores = (float*)(ws + alloc((size_t)MROWS * 64 * 4));
    float* Pmat   = (float*)(ws + alloc((size_t)BB * MCLUS * NPAT * 4));
    float* cam_x  = (float*)(ws + alloc((size_t)16 * CC * 4));
    float* cam_h1 = (float*)(ws + alloc((size_t)16 * 1024 * 4));
    float* cam_h2 = (float*)(ws + alloc((size_t)16 * 512 * 4));
    float* cam_g  = (float*)(ws + alloc((size_t)4 * 512 * 4));
    float* cam_o1 = (float*)(ws + alloc((size_t)4 * 512 * 4));
    float* cam_o2 = (float*)(ws + alloc((size_t)4 * 256 * 4));
    float* reg_x  = (float*)(ws + alloc((size_t)64 * CC * 4));
    float* reg_h1 = (float*)(ws + alloc((size_t)64 * 1024 * 4));
    float* reg_h2 = (float*)(ws + alloc((size_t)64 * 512 * 4));
    float* reg_g  = (float*)(ws + alloc((size_t)4 * 512 * 4));
    float* reg_o1 = (float*)(ws + alloc((size_t)4 * 512 * 4));
    float* reg_o2 = (float*)(ws + alloc((size_t)4 * 256 * 4));

    // --- pack & convert ---
    pack_patch<<<(int)(((size_t)MROWS * CC / 8 + 255) / 256), 256, 0, stream>>>(tokens, A_bf);
    transpose_w_bf16<<<(2048 * 512 + 255) / 256, 256, 0, stream>>>(pf_w1, W1t, 2048, 512);
    transpose_w_bf16<<<(2048 * 512 + 255) / 256, 256, 0, stream>>>(ps_w1, W1t + (size_t)512 * 2048, 2048, 512);
    transpose_w_bf16<<<(512 * 128 + 255) / 256, 256, 0, stream>>>(pf_w2, W2ft, 512, 128);
    transpose_w_bf16<<<(512 * 64 + 255) / 256, 256, 0, stream>>>(ps_w2, W2st, 512, 64);

    // --- big WMMA GEMMs ---
    // H = relu(patch @ [pf_w1|ps_w1] + [pf_b1|ps_b1])  [16384,1024] bf16
    gemm_bf16_wmma<true, true><<<1024, 256, 0, stream>>>(
        A_bf, 2048, W1t, 2048, Hbf, 1024, MROWS, 1024, 2048, pf_b1, ps_b1, 512);
    // feats = H[:, :512] @ pf_w2 + pf_b2   [16384,128] f32
    gemm_bf16_wmma<false, false><<<128, 256, 0, stream>>>(
        Hbf, 1024, W2ft, 512, feats, 128, MROWS, 128, 512, pf_b2, pf_b2, 1 << 30);
    // scores = H[:, 512:] @ ps_w2 + ps_b2  [16384,64] f32
    gemm_bf16_wmma<false, false><<<64, 256, 0, stream>>>(
        Hbf + 512, 1024, W2st, 512, scores, 64, MROWS, 64, 512, ps_b2, ps_b2, 1 << 30);

    // --- camera token pooling (L = 4, rows = 16) ---
    gather_tokens<<<(16 * CC + 255) / 256, 256, 0, stream>>>(tokens, cam_x, 0, 1);
    linear_f32<<<(16 * 1024 + 255) / 256, 256, 0, stream>>>(cam_x, cam_w1, cam_b1, cam_h1, 16, 1024, 2048, 1);
    linear_f32<<<(16 * 512 + 255) / 256, 256, 0, stream>>>(cam_h1, cam_w2, cam_b2, cam_h2, 16, 512, 1024, 0);
    l2norm_rows<<<16, 256, 0, stream>>>(cam_h2, cam_h2, 512, 512, 0);
    gem_pool<<<(4 * 512 + 255) / 256, 256, 0, stream>>>(cam_h2, cam_g, 4, 512, cam_p);
    linear_f32<<<(4 * 512 + 255) / 256, 256, 0, stream>>>(cam_g, cam_w3, cam_b3, cam_o1, 4, 512, 512, 1);
    linear_f32<<<(4 * 256 + 255) / 256, 256, 0, stream>>>(cam_o1, cam_w4, cam_b4, cam_o2, 4, 256, 512, 0);
    l2norm_rows<<<4, 256, 0, stream>>>(cam_o2, out, 256, OUTROW, 8192);

    // --- register token pooling (L = 16, rows = 64) ---
    gather_tokens<<<(64 * CC + 255) / 256, 256, 0, stream>>>(tokens, reg_x, 1, 4);
    linear_f32<<<(64 * 1024 + 255) / 256, 256, 0, stream>>>(reg_x, reg_w1, reg_b1, reg_h1, 64, 1024, 2048, 1);
    linear_f32<<<(64 * 512 + 255) / 256, 256, 0, stream>>>(reg_h1, reg_w2, reg_b2, reg_h2, 64, 512, 1024, 0);
    l2norm_rows<<<64, 256, 0, stream>>>(reg_h2, reg_h2, 512, 512, 0);
    gem_pool<<<(4 * 512 + 255) / 256, 256, 0, stream>>>(reg_h2, reg_g, 16, 512, reg_p);
    linear_f32<<<(4 * 512 + 255) / 256, 256, 0, stream>>>(reg_g, reg_w3, reg_b3, reg_o1, 4, 512, 512, 1);
    linear_f32<<<(4 * 256 + 255) / 256, 256, 0, stream>>>(reg_o1, reg_w4, reg_b4, reg_o2, 4, 256, 512, 0);
    l2norm_rows<<<4, 256, 0, stream>>>(reg_o2, out, 256, OUTROW, 8448);

    // --- sinkhorn + cluster pooling ---
    sinkhorn_kernel<<<BB, 1024, 0, stream>>>(scores, dust, Pmat);
    pooled_norm<<<BB * MCLUS, 128, 0, stream>>>(feats, Pmat, out);
}
